// Decoder_81810537055416
// MI455X (gfx1250) — matrix operand order repaired and checked by reading, not compiled, not run on hardware
//
#include <hip/hip_runtime.h>
#include <hip/hip_bf16.h>

typedef __attribute__((ext_vector_type(16))) _Float16 v16h;
typedef __attribute__((ext_vector_type(8)))  float    v8f;

#define B_SZ 256
#define TENC 200
#define TD   200
#define INR  400   // NMEL * R
#define TMELSTRIDE 80000  // T_MEL * NMEL

// ---------------- utility kernels ----------------

__global__ void k_zero(float* p, int n) {
  int i = blockIdx.x * blockDim.x + threadIdx.x;
  if (i < n) p[i] = 0.f;
}

// convert f32 weight [N,K] row-major -> f16 [N,Kp] zero-padded in K
__global__ void k_cvt_w(const float* __restrict__ src, _Float16* __restrict__ dst,
                        int N, int K, int Kp) {
  int i = blockIdx.x * blockDim.x + threadIdx.x;
  if (i >= N * Kp) return;
  int n = i / Kp, k = i - n * Kp;
  dst[i] = (k < K) ? (_Float16)src[n * K + k] : (_Float16)0.0f;
}

// out[B, na+nb] = concat(a[B,na], b[B,nb])
__global__ void k_concat(const float* __restrict__ a, int na,
                         const float* __restrict__ b, int nb,
                         float* __restrict__ out) {
  int i = blockIdx.x * blockDim.x + threadIdx.x;
  int w = na + nb;
  if (i >= B_SZ * w) return;
  int r = i / w, c = i - r * w;
  out[i] = (c < na) ? a[r * na + c] : b[r * nb + (c - na)];
}

// GRU gate combine: h = (1-z)*n + z*h ; optional residual resout = h + res
__global__ void k_gru(const float* __restrict__ gi, const float* __restrict__ gh,
                      float* __restrict__ h, const float* __restrict__ res,
                      float* __restrict__ resout) {
  int i = blockIdx.x * blockDim.x + threadIdx.x;  // over B*256
  int b = i >> 8, j = i & 255;
  const float* gib = gi + b * 768;
  const float* ghb = gh + b * 768;
  float r = 1.f / (1.f + expf(-(gib[j]       + ghb[j])));
  float z = 1.f / (1.f + expf(-(gib[256 + j] + ghb[256 + j])));
  float n = tanhf(gib[512 + j] + r * ghb[512 + j]);
  float hn = (1.f - z) * n + z * h[i];
  h[i] = hn;
  if (res) resout[i] = hn + res[i];
}

// ---------------- WMMA GEMM: C[M,N] = act(A[M,K] @ W[N,K]^T + bias) ----------------
// 4 waves per block; each wave computes one 16x16 tile via v_wmma_f32_16x16x32_f16.
// A: f32, row stride lda; rows are readable for ceil32(K) when Kp==K, or the
// tail (Kp = K+16, K%32==16) is handled with a statically zeroed upper half.
// W: f16 row-major [N,Kp], Kp multiple of 32, zero-padded in K.
__global__ void __launch_bounds__(128)
k_gemm(const float* __restrict__ A, int lda, int K,
       const _Float16* __restrict__ W, int Kp,
       const float* __restrict__ bias,
       float* __restrict__ C, int ldc, int N, int act) {
  const int lane  = threadIdx.x & 31;
  const int wave  = threadIdx.x >> 5;
  const int ntile = N >> 4;
  const int tile  = (blockIdx.x << 2) + wave;
  if (tile >= ntile) return;              // uniform per wave: EXEC stays all-ones
  const int rowbase = blockIdx.y << 4;
  const int colbase = tile << 4;
  const int mrow = rowbase + (lane & 15);
  const int ncol = colbase + (lane & 15);
  const int kA   = (lane < 16) ? 0 : 8;   // A: lanes 0-15 K 0-7/16-23, 16-31 K 8-15/24-31
  const int kB   = (lane < 16) ? 0 : 8;   // B: lanes 0-15 K 0-7/16-23, 16-31 K 8-15/24-31
  const float*    Ap = A + (long)mrow * lda + kA;
  const _Float16* Wp = W + (long)ncol * Kp + kB;
  v8f c = {0.f, 0.f, 0.f, 0.f, 0.f, 0.f, 0.f, 0.f};
  const int kfull = (Kp == K) ? Kp : (Kp - 32);
  for (int kk = 0; kk < kfull; kk += 32) {
    const float4* p0 = (const float4*)(Ap + kk);       // 32B aligned
    const float4* p1 = (const float4*)(Ap + kk + 16);
    float4 f0 = p0[0], f1 = p0[1];
    float4 f2 = p1[0], f3 = p1[1];
    v16h a;
    a[0]  = (_Float16)f0.x; a[1]  = (_Float16)f0.y;
    a[2]  = (_Float16)f0.z; a[3]  = (_Float16)f0.w;
    a[4]  = (_Float16)f1.x; a[5]  = (_Float16)f1.y;
    a[6]  = (_Float16)f1.z; a[7]  = (_Float16)f1.w;
    a[8]  = (_Float16)f2.x; a[9]  = (_Float16)f2.y;
    a[10] = (_Float16)f2.z; a[11] = (_Float16)f2.w;
    a[12] = (_Float16)f3.x; a[13] = (_Float16)f3.y;
    a[14] = (_Float16)f3.z; a[15] = (_Float16)f3.w;
    const _Float16* wrow = Wp + kk;                    // K 0-7 and 16-23 relative to kB
    v16h b;
#pragma unroll
    for (int v = 0; v < 16; ++v) b[v] = wrow[(v < 8) ? v : v + 8];
    c = __builtin_amdgcn_wmma_f32_16x16x32_f16(false, a, false, b, (short)0, c,
                                               false, false);
  }
  if (Kp != K) {  // tail chunk: real K values only in the lower half of A regs
    const float4* p0 = (const float4*)(Ap + kfull);
    float4 f0 = p0[0], f1 = p0[1];
    v16h a;
    a[0] = (_Float16)f0.x; a[1] = (_Float16)f0.y;
    a[2] = (_Float16)f0.z; a[3] = (_Float16)f0.w;
    a[4] = (_Float16)f1.x; a[5] = (_Float16)f1.y;
    a[6] = (_Float16)f1.z; a[7] = (_Float16)f1.w;
#pragma unroll
    for (int v = 8; v < 16; ++v) a[v] = (_Float16)0.0f;
    const _Float16* wrow = Wp + kfull;                 // W is zero-padded
    v16h b;
#pragma unroll
    for (int v = 0; v < 16; ++v) b[v] = wrow[(v < 8) ? v : v + 8];
    c = __builtin_amdgcn_wmma_f32_16x16x32_f16(false, a, false, b, (short)0, c,
                                               false, false);
  }
  float bv = bias ? bias[ncol] : 0.f;
  int mb = rowbase + ((lane < 16) ? 0 : 8);  // C/D: VGPR j -> M=j (lo) / j+8 (hi)
#pragma unroll
  for (int j = 0; j < 8; ++j) {
    float v = c[j] + bv;
    if (act) v = fmaxf(v, 0.f);
    C[(long)(mb + j) * ldc + ncol] = v;
  }
}

// ---------------- Bahdanau attention (one block per batch element) ----------------
__global__ void k_attn(const float* __restrict__ pm,    // [B,TENC,256]
                       const float* __restrict__ q,     // [B,256]
                       const float* __restrict__ enc,   // [B,TENC,256]
                       const float* __restrict__ vW,    // [256]
                       const int*   __restrict__ mlen,  // [B]
                       float* __restrict__ ctx,         // [B,256]
                       float* __restrict__ alignOut) {  // base + t*TENC; row stride TD*TENC
  __shared__ float sq[256];
  __shared__ float sal[256];
  __shared__ float red[256];
  const int b    = blockIdx.x;
  const int tid  = threadIdx.x;
  const int lane = tid & 31;
  const int wave = tid >> 5;
  sq[tid] = q[b * 256 + tid];
  __syncthreads();
  const float* pmb = pm + (long)b * TENC * 256;
  const int L = mlen[b];
  for (int t = wave; t < TENC; t += 8) {
    const float* row = pmb + t * 256;
    float p = 0.f;
    for (int k = lane; k < 256; k += 32)
      p += tanhf(row[k] + sq[k]) * vW[k];
    for (int off = 16; off > 0; off >>= 1)
      p += __shfl_xor(p, off, 32);
    if (lane == 0) sal[t] = (t >= L) ? -1e9f : p;
  }
  __syncthreads();
  // softmax over TENC
  red[tid] = (tid < TENC) ? sal[tid] : -3.4e38f;
  __syncthreads();
  for (int s = 128; s > 0; s >>= 1) {
    if (tid < s) red[tid] = fmaxf(red[tid], red[tid + s]);
    __syncthreads();
  }
  float mx = red[0];
  __syncthreads();
  float e = (tid < TENC) ? expf(sal[tid] - mx) : 0.f;
  red[tid] = e;
  __syncthreads();
  for (int s = 128; s > 0; s >>= 1) {
    if (tid < s) red[tid] += red[tid + s];
    __syncthreads();
  }
  float inv = 1.f / red[0];
  __syncthreads();
  if (tid < TENC) {
    float a = e * inv;
    sal[tid] = a;
    alignOut[(long)b * (TD * TENC) + tid] = a;
  }
  __syncthreads();
  const float* encb = enc + (long)b * TENC * 256;
  float acc = 0.f;
  for (int t = 0; t < TENC; ++t)
    acc += sal[t] * encb[t * 256 + tid];
  ctx[b * 256 + tid] = acc;
}

// ---------------- host launcher ----------------
extern "C" void kernel_launch(void* const* d_in, const int* in_sizes, int n_in,
                              void* d_out, int out_size, void* d_ws, size_t ws_size,
                              hipStream_t stream) {
  (void)in_sizes; (void)n_in; (void)out_size; (void)ws_size;
  const float* enc   = (const float*)d_in[0];
  const float* inp   = (const float*)d_in[1];
  const int*   mlen  = (const int*)d_in[2];
  const float* W1    = (const float*)d_in[3];
  const float* b1    = (const float*)d_in[4];
  const float* W2    = (const float*)d_in[5];
  const float* b2    = (const float*)d_in[6];
  const float* aWih  = (const float*)d_in[7];
  const float* abih  = (const float*)d_in[8];
  const float* aWhh  = (const float*)d_in[9];
  const float* abhh  = (const float*)d_in[10];
  const float* memW  = (const float*)d_in[11];
  const float* qW    = (const float*)d_in[12];
  const float* vW    = (const float*)d_in[13];
  const float* pW    = (const float*)d_in[14];
  const float* pb    = (const float*)d_in[15];
  const float* d1Wih = (const float*)d_in[16];
  const float* d1bih = (const float*)d_in[17];
  const float* d1Whh = (const float*)d_in[18];
  const float* d1bhh = (const float*)d_in[19];
  const float* d2Wih = (const float*)d_in[20];
  const float* d2bih = (const float*)d_in[21];
  const float* d2Whh = (const float*)d_in[22];
  const float* d2bhh = (const float*)d_in[23];
  const float* melW  = (const float*)d_in[24];
  const float* melb  = (const float*)d_in[25];

  // ---- workspace layout (f32 region, then f16 weight region) ----
  float* fw = (float*)d_ws;
  size_t off = 0;
  auto allocF = [&](size_t n) { float* p = fw + off; off += n; return p; };
  float* pmem   = allocF((size_t)B_SZ * TENC * 256);  // processed_memory (52 MB, L2-resident)
  float* attn_h = allocF((size_t)B_SZ * 256);
  float* h1     = allocF((size_t)B_SZ * 256);
  float* h2     = allocF((size_t)B_SZ * 256);
  float* ctx    = allocF((size_t)B_SZ * 256);
  float* x1     = allocF((size_t)B_SZ * 256);
  float* x2     = allocF((size_t)B_SZ * 128);
  float* cellin = allocF((size_t)B_SZ * 384);
  float* gi     = allocF((size_t)B_SZ * 768);
  float* gh     = allocF((size_t)B_SZ * 768);
  float* qv     = allocF((size_t)B_SZ * 256);
  float* dcat   = allocF((size_t)B_SZ * 512);
  float* dvec   = allocF((size_t)B_SZ * 256);
  float* zprev  = allocF((size_t)B_SZ * INR);

  _Float16* hw = (_Float16*)(fw + off);
  size_t hoff = 0;
  auto allocH = [&](size_t n) { _Float16* p = hw + hoff; hoff += n; return p; };
  _Float16* W1h    = allocH(256 * 416);
  _Float16* W2h    = allocH(128 * 256);
  _Float16* aWihH  = allocH(768 * 384);
  _Float16* aWhhH  = allocH(768 * 256);
  _Float16* memWh  = allocH(256 * 256);
  _Float16* qWh    = allocH(256 * 256);
  _Float16* pWh    = allocH(256 * 512);
  _Float16* d1WihH = allocH(768 * 256);
  _Float16* d1WhhH = allocH(768 * 256);
  _Float16* d2WihH = allocH(768 * 256);
  _Float16* d2WhhH = allocH(768 * 256);
  _Float16* melWh  = allocH(400 * 256);

  auto cvt = [&](const float* s, _Float16* d, int N, int K, int Kp) {
    int n = N * Kp;
    k_cvt_w<<<(n + 255) / 256, 256, 0, stream>>>(s, d, N, K, Kp);
  };
  cvt(W1, W1h, 256, 400, 416);
  cvt(W2, W2h, 128, 256, 256);
  cvt(aWih, aWihH, 768, 384, 384);
  cvt(aWhh, aWhhH, 768, 256, 256);
  cvt(memW, memWh, 256, 256, 256);
  cvt(qW, qWh, 256, 256, 256);
  cvt(pW, pWh, 256, 512, 512);
  cvt(d1Wih, d1WihH, 768, 256, 256);
  cvt(d1Whh, d1WhhH, 768, 256, 256);
  cvt(d2Wih, d2WihH, 768, 256, 256);
  cvt(d2Whh, d2WhhH, 768, 256, 256);
  cvt(melW, melWh, 400, 256, 256);

  auto zero = [&](float* p, int n) {
    k_zero<<<(n + 255) / 256, 256, 0, stream>>>(p, n);
  };
  zero(attn_h, B_SZ * 256);
  zero(h1, B_SZ * 256);
  zero(h2, B_SZ * 256);
  zero(ctx, B_SZ * 256);
  zero(zprev, B_SZ * INR);

  auto gemm = [&](const float* A, int lda, int K, const _Float16* Wh, int Kp,
                  const float* bias, float* C, int ldc, int M, int N, int act) {
    int ntile = N / 16;
    k_gemm<<<dim3((ntile + 3) / 4, M / 16), 128, 0, stream>>>(A, lda, K, Wh, Kp,
                                                              bias, C, ldc, N, act);
  };

  // one-time: processed_memory = encoder_outputs @ mem_W^T   [B*TENC, 256]
  gemm(enc, 256, 256, memWh, 256, nullptr, pmem, 256, B_SZ * TENC, 256, 0);

  float* outputs = (float*)d_out;
  float* aligns  = outputs + (size_t)B_SZ * TD * INR;

  for (int t = 0; t < TD; ++t) {
    // teacher-forced input: zeros at t=0, else inputs_r[:, t-1, :]
    const float* prevA = (t == 0) ? zprev : (inp + (size_t)(t - 1) * INR);
    int lda = (t == 0) ? INR : TMELSTRIDE;
    // prenet
    gemm(prevA, lda, 400, W1h, 416, b1, x1, 256, B_SZ, 256, 1);
    gemm(x1, 256, 256, W2h, 256, b2, x2, 128, B_SZ, 128, 1);
    // attention GRU
    k_concat<<<(B_SZ * 384 + 255) / 256, 256, 0, stream>>>(x2, 128, ctx, 256, cellin);
    gemm(cellin, 384, 384, aWihH, 384, abih, gi, 768, B_SZ, 768, 0);
    gemm(attn_h, 256, 256, aWhhH, 256, abhh, gh, 768, B_SZ, 768, 0);
    k_gru<<<B_SZ, 256, 0, stream>>>(gi, gh, attn_h, nullptr, nullptr);
    // Bahdanau attention
    gemm(attn_h, 256, 256, qWh, 256, nullptr, qv, 256, B_SZ, 256, 0);
    k_attn<<<B_SZ, 256, 0, stream>>>(pmem, qv, enc, vW, mlen, ctx,
                                     aligns + (size_t)t * TENC);
    // project to decoder input
    k_concat<<<(B_SZ * 512 + 255) / 256, 256, 0, stream>>>(attn_h, 256, ctx, 256, dcat);
    gemm(dcat, 512, 512, pWh, 512, pb, dvec, 256, B_SZ, 256, 0);
    // decoder GRU 1 (+ residual)
    gemm(dvec, 256, 256, d1WihH, 256, d1bih, gi, 768, B_SZ, 768, 0);
    gemm(h1, 256, 256, d1WhhH, 256, d1bhh, gh, 768, B_SZ, 768, 0);
    k_gru<<<B_SZ, 256, 0, stream>>>(gi, gh, h1, dvec, dvec);
    // decoder GRU 2 (+ residual)
    gemm(dvec, 256, 256, d2WihH, 256, d2bih, gi, 768, B_SZ, 768, 0);
    gemm(h2, 256, 256, d2WhhH, 256, d2bhh, gh, 768, B_SZ, 768, 0);
    k_gru<<<B_SZ, 256, 0, stream>>>(gi, gh, h2, dvec, dvec);
    // project to mel, write straight into outputs[:, t, :]
    gemm(dvec, 256, 256, melWh, 256, melb, outputs + (size_t)t * INR,
         TD * INR, B_SZ, 400, 0);
  }
}
